// GNN_MDN_44779329028529
// MI455X (gfx1250) — matrix operand order repaired and checked by reading, not compiled, not run on hardware
//
#include <hip/hip_runtime.h>
#include <hip/hip_bf16.h>

typedef __attribute__((ext_vector_type(16))) _Float16 v16h;
typedef __attribute__((ext_vector_type(8)))  _Float16 v8h;
typedef __attribute__((ext_vector_type(4)))  _Float16 v4h;
typedef __attribute__((ext_vector_type(8)))  float    v8f;
typedef __attribute__((ext_vector_type(4)))  float    v4f;

#define NNODES 30000
#define NEDGES 480000
#define ETOT   (NEDGES + NNODES)   // edges + self loops = 510000
#define EMB    64
#define HID    32
#define NH1    8
#define F1     (NH1 * HID)         // 256
#define NMIX   5
#define CDIV(a, b) (((a) + (b) - 1) / (b))

static __device__ __forceinline__ float elu1(float v) {
    return v > 0.f ? v : (__expf(v) - 1.f);
}

static __device__ __forceinline__ void atomicMaxF(float* addr, float val) {
    unsigned* ua = reinterpret_cast<unsigned*>(addr);
    unsigned old = *ua;
    while (__uint_as_float(old) < val) {
        unsigned assumed = old;
        old = atomicCAS(ua, assumed, __float_as_uint(val));
        if (old == assumed) break;
    }
}

// ---------------- utility kernels ----------------

__global__ void k_fill(float* p, float v, int n) {
    int i = blockIdx.x * blockDim.x + threadIdx.x;
    if (i < n) p[i] = v;
}

// W[K,N] f32  ->  Wt[N,K] f16  (tiny, one-off per layer)
__global__ void k_transpose_to_f16(const float* __restrict__ S, _Float16* __restrict__ D,
                                   int K, int N) {
    int i = blockIdx.x * blockDim.x + threadIdx.x;
    if (i >= K * N) return;
    int k = i / N, n = i % N;
    D[(size_t)n * K + k] = (_Float16)S[i];
}

// x16[n, 0:64] = (f16) embedding[node_ids[n], 0:64]; float4 in, 4 halves out
__global__ void k_gather_f16(const int* __restrict__ ids, const float* __restrict__ emb,
                             _Float16* __restrict__ x, int nq) {
    int i = blockIdx.x * blockDim.x + threadIdx.x;   // quad index
    if (i >= nq) return;
    int node = i >> 4;                 // / (EMB/4)
    int q    = (i & 15) << 2;          // element offset within row
    v4f e = *(const v4f*)(emb + (size_t)ids[node] * EMB + q);
    v4h h;
    h[0] = (_Float16)e[0]; h[1] = (_Float16)e[1];
    h[2] = (_Float16)e[2]; h[3] = (_Float16)e[3];
    *(v4h*)(x + (size_t)node * EMB + q) = h;
}

// ---------------- WMMA GEMM: C[M,N] = A[M,K] * Bt[N,K]^T, f16 in / f32 out ----------------
// One wave per 16x16 output tile, K unrolled at compile time.
// Fragment layouts per CDNA5 ISA 7.12.2 (wave32):
//  A (16x32, 16-bit): lane M=lane&15; halves 0-7 = K base+0..7, halves 8-15 = K base+16..23,
//                     base += 8 for lanes 16-31  -> two contiguous 16B loads.
//  B (32x16, 16-bit): lane N=lane&15; halves 0-7 = K base+0..7, halves 8-15 = K base+16..23, same as A.
//                     With B pre-transposed to [N,K], that is two contiguous 16B loads.
//  C/D (16x16 f32):   VGPR r -> M = r + 8*(lane>=16), N = lane&15.

template <int K>
__global__ void k_gemm_wmma_f16(const _Float16* __restrict__ A,
                                const _Float16* __restrict__ Bt,
                                float* __restrict__ C,
                                int N, int totalTiles) {
    int wave = threadIdx.x >> 5;
    int tile = blockIdx.x * (blockDim.x >> 5) + wave;
    if (tile >= totalTiles) return;              // wave-uniform: EXEC stays all-ones for WMMA
    int ntiles = N >> 4;
    int mt = tile / ntiles;
    int nt = tile % ntiles;
    int lane = threadIdx.x & 31;
    int hi   = lane >> 4;                        // half-wave select
    int lcol = lane & 15;

    int arow = mt * 16 + lcol;
    int bcol = nt * 16 + lcol;
    const _Float16* aptr = A  + (size_t)arow * K + (hi ? 8 : 0);
    const _Float16* bptr = Bt + (size_t)bcol * K + (hi ? 8 : 0);

    v8f c = {};
#pragma unroll
    for (int kb = 0; kb < K; kb += 32) {
        v8h a0 = *(const v8h*)(aptr + kb);        // K = kb+base+0..7
        v8h a1 = *(const v8h*)(aptr + kb + 16);   // K = kb+base+16..23
        v8h b0 = *(const v8h*)(bptr + kb);        // K = kb+base+0..7
        v8h b1 = *(const v8h*)(bptr + kb + 16);   // K = kb+base+16..23
        v16h a = __builtin_shufflevector(a0, a1, 0,1,2,3,4,5,6,7,8,9,10,11,12,13,14,15);
        v16h b = __builtin_shufflevector(b0, b1, 0,1,2,3,4,5,6,7,8,9,10,11,12,13,14,15);
        c = __builtin_amdgcn_wmma_f32_16x16x32_f16(
                /*neg_a=*/false, a, /*neg_b=*/false, b,
                /*c_mod=*/(short)0, c, /*reuse_a=*/false, /*reuse_b=*/false);
    }
    int rbase = mt * 16 + (hi << 3);
#pragma unroll
    for (int r = 0; r < 8; ++r)
        C[(size_t)(rbase + r) * N + nt * 16 + lcol] = c[r];
}

// ---------------- attention dot products: al_s[n,h], al_d[n,h] ----------------

__global__ void k_att_dots(const float* __restrict__ h, const float* __restrict__ asrc,
                           const float* __restrict__ adst, float* __restrict__ als,
                           float* __restrict__ ald, int heads) {
    int t = blockIdx.x * blockDim.x + threadIdx.x;
    if (t >= NNODES * heads) return;
    int n = t / heads, hd = t % heads;
    const float* hp = h + (size_t)n * heads * HID + hd * HID;
    const float* as = asrc + hd * HID;
    const float* ad = adst + hd * HID;
    float s = 0.f, d = 0.f;
#pragma unroll
    for (int c = 0; c < HID; ++c) { s += hp[c] * as[c]; d += hp[c] * ad[c]; }
    als[t] = s;
    ald[t] = d;
}

// ---------------- edge passes (segment softmax + aggregation) ----------------

__global__ void k_edge_lrelu_max(const int* __restrict__ src, const int* __restrict__ dst,
                                 const float* __restrict__ als, const float* __restrict__ ald,
                                 float* __restrict__ e, float* __restrict__ m, int heads) {
    int t = blockIdx.x * blockDim.x + threadIdx.x;
    if (t >= ETOT * heads) return;
    int ed = t / heads, hd = t % heads;
    int s = ed < NEDGES ? src[ed] : ed - NEDGES;   // self-loop tail
    int d = ed < NEDGES ? dst[ed] : ed - NEDGES;
    float v = als[s * heads + hd] + ald[d * heads + hd];
    v = v > 0.f ? v : 0.2f * v;                    // leaky_relu(0.2)
    e[t] = v;
    atomicMaxF(&m[d * heads + hd], v);
}

__global__ void k_edge_expsum(const int* __restrict__ dst,
                              float* __restrict__ e, const float* __restrict__ m,
                              float* __restrict__ ssum, int heads) {
    int t = blockIdx.x * blockDim.x + threadIdx.x;
    if (t >= ETOT * heads) return;
    int ed = t / heads, hd = t % heads;
    int d = ed < NEDGES ? dst[ed] : ed - NEDGES;
    float w = __expf(e[t] - m[d * heads + hd]);
    e[t] = w;
    atomicAdd(&ssum[d * heads + hd], w);
}

__global__ void k_edge_aggregate(const int* __restrict__ src, const int* __restrict__ dst,
                                 const float* __restrict__ e, const float* __restrict__ ssum,
                                 const float* __restrict__ h, float* __restrict__ out, int heads) {
    int t = blockIdx.x * blockDim.x + threadIdx.x;
    if (t >= ETOT * heads) return;
    int ed = t / heads, hd = t % heads;
    int s = ed < NEDGES ? src[ed] : ed - NEDGES;
    int d = ed < NEDGES ? dst[ed] : ed - NEDGES;
    float alpha = e[t] / ssum[d * heads + hd];
    const float* hs = h + (size_t)s * heads * HID + hd * HID;
    float* od = out + (size_t)d * heads * HID + hd * HID;
#pragma unroll
    for (int c = 0; c < HID; ++c)
        atomicAdd(&od[c], hs[c] * alpha);
}

// ---------------- bias + ELU finalizers ----------------

__global__ void k_bias_elu_to_f16(const float* __restrict__ acc, const float* __restrict__ bias,
                                  _Float16* __restrict__ xo, int feat, int n) {
    int i = blockIdx.x * blockDim.x + threadIdx.x;
    if (i >= n) return;
    xo[i] = (_Float16)elu1(acc[i] + bias[i % feat]);
}

__global__ void k_bias_elu_f32(const float* __restrict__ acc, const float* __restrict__ bias,
                               float* __restrict__ xo, int feat, int n) {
    int i = blockIdx.x * blockDim.x + threadIdx.x;
    if (i >= n) return;
    xo[i] = elu1(acc[i] + bias[i % feat]);
}

// ---------------- MDN head: fc(32x32)+ELU, then mu / softplus(var) / softmax(pi) ----------------

__global__ void k_head(const float* __restrict__ x2,
                       const float* __restrict__ Wfc, const float* __restrict__ bfc,
                       const float* __restrict__ Wmu, const float* __restrict__ bmu,
                       const float* __restrict__ Wvar, const float* __restrict__ bvar,
                       const float* __restrict__ Wpi, const float* __restrict__ bpi,
                       float* __restrict__ out) {
    int n = blockIdx.x * blockDim.x + threadIdx.x;
    if (n >= NNODES) return;
    float xi[HID], y[HID];
#pragma unroll
    for (int k = 0; k < HID; ++k) xi[k] = x2[(size_t)n * HID + k];
    for (int j = 0; j < HID; ++j) {
        float a = bfc[j];
#pragma unroll
        for (int k = 0; k < HID; ++k) a += xi[k] * Wfc[k * HID + j];
        y[j] = elu1(a);
    }
    float mu[NMIX], vr[NMIX], pi[NMIX];
    for (int m = 0; m < NMIX; ++m) {
        float am = bmu[m], av = bvar[m], ap = bpi[m];
#pragma unroll
        for (int j = 0; j < HID; ++j) {
            float yj = y[j];
            am += yj * Wmu[j * NMIX + m];
            av += yj * Wvar[j * NMIX + m];
            ap += yj * Wpi[j * NMIX + m];
        }
        mu[m] = am;
        vr[m] = av > 20.f ? av : log1pf(__expf(av));  // softplus
        pi[m] = ap;
    }
    float pm = pi[0];
    for (int m = 1; m < NMIX; ++m) pm = fmaxf(pm, pi[m]);
    float ps = 0.f;
    for (int m = 0; m < NMIX; ++m) { pi[m] = __expf(pi[m] - pm); ps += pi[m]; }
    float inv = 1.f / ps;
    float* omu  = out + (size_t)n * NMIX;
    float* ovar = out + (size_t)NNODES * NMIX + (size_t)n * NMIX;
    float* opi  = out + (size_t)2 * NNODES * NMIX + (size_t)n * NMIX;
    for (int m = 0; m < NMIX; ++m) {
        omu[m]  = mu[m];
        ovar[m] = vr[m];
        opi[m]  = pi[m] * inv;
    }
}

// ---------------- launcher ----------------

extern "C" void kernel_launch(void* const* d_in, const int* in_sizes, int n_in,
                              void* d_out, int out_size, void* d_ws, size_t ws_size,
                              hipStream_t stream) {
    (void)in_sizes; (void)n_in; (void)out_size; (void)ws_size;
    const int*   node_ids = (const int*)d_in[0];
    const int*   edge     = (const int*)d_in[1];
    const int*   src      = edge;
    const int*   dst      = edge + NEDGES;
    const float* emb  = (const float*)d_in[2];
    const float* W1   = (const float*)d_in[3];
    const float* as1  = (const float*)d_in[4];
    const float* ad1  = (const float*)d_in[5];
    const float* b1   = (const float*)d_in[6];
    const float* W2   = (const float*)d_in[7];
    const float* as2  = (const float*)d_in[8];
    const float* ad2  = (const float*)d_in[9];
    const float* b2   = (const float*)d_in[10];
    const float* Wfc  = (const float*)d_in[11];
    const float* bfc  = (const float*)d_in[12];
    const float* Wmu  = (const float*)d_in[13];
    const float* bmu  = (const float*)d_in[14];
    const float* Wvar = (const float*)d_in[15];
    const float* bvar = (const float*)d_in[16];
    const float* Wpi  = (const float*)d_in[17];
    const float* bpi  = (const float*)d_in[18];
    float* out = (float*)d_out;

    char*  w   = (char*)d_ws;
    size_t off = 0;
    auto carve = [&](size_t bytes) -> void* {
        void* p = w + off;
        off += (bytes + 255) & ~(size_t)255;
        return p;
    };

    _Float16* x16  = (_Float16*)carve((size_t)NNODES * EMB * 2);
    _Float16* w1t  = (_Float16*)carve((size_t)EMB * F1 * 2);     // W1^T [256,64] f16
    float*    h1   = (float*)carve((size_t)NNODES * F1 * 4);
    float*    als1 = (float*)carve((size_t)NNODES * NH1 * 4);
    float*    ald1 = (float*)carve((size_t)NNODES * NH1 * 4);
    float*    m1   = (float*)carve((size_t)NNODES * NH1 * 4);
    float*    s1   = (float*)carve((size_t)NNODES * NH1 * 4);
    float*    e1   = (float*)carve((size_t)ETOT * NH1 * 4);
    float*    out1 = (float*)carve((size_t)NNODES * F1 * 4);
    _Float16* x1h  = (_Float16*)carve((size_t)NNODES * F1 * 2);
    _Float16* w2t  = (_Float16*)carve((size_t)F1 * HID * 2);     // W2^T [32,256] f16
    float*    h2   = (float*)carve((size_t)NNODES * HID * 4);
    float*    als2 = (float*)carve((size_t)NNODES * 4);
    float*    ald2 = (float*)carve((size_t)NNODES * 4);
    float*    m2   = (float*)carve((size_t)NNODES * 4);
    float*    s2   = (float*)carve((size_t)NNODES * 4);
    float*    e2   = (float*)carve((size_t)ETOT * 4);
    float*    out2 = (float*)carve((size_t)NNODES * HID * 4);
    float*    x2   = (float*)carve((size_t)NNODES * HID * 4);

    const int B = 256;
    const float NEG = -3.0e38f;

    // ---- init accumulators (ws is poisoned; must re-init every call) ----
    k_fill<<<CDIV(NNODES * NH1, B), B, 0, stream>>>(m1, NEG, NNODES * NH1);
    k_fill<<<CDIV(NNODES * NH1, B), B, 0, stream>>>(s1, 0.f, NNODES * NH1);
    k_fill<<<CDIV(NNODES * F1, B), B, 0, stream>>>(out1, 0.f, NNODES * F1);
    k_fill<<<CDIV(NNODES, B), B, 0, stream>>>(m2, NEG, NNODES);
    k_fill<<<CDIV(NNODES, B), B, 0, stream>>>(s2, 0.f, NNODES);
    k_fill<<<CDIV(NNODES * HID, B), B, 0, stream>>>(out2, 0.f, NNODES * HID);

    // ---- embedding gather + weight transpose/convert to f16 ----
    k_gather_f16<<<CDIV(NNODES * (EMB / 4), B), B, 0, stream>>>(node_ids, emb, x16,
                                                                NNODES * (EMB / 4));
    k_transpose_to_f16<<<CDIV(EMB * F1, B), B, 0, stream>>>(W1, w1t, EMB, F1);
    k_transpose_to_f16<<<CDIV(F1 * HID, B), B, 0, stream>>>(W2, w2t, F1, HID);

    // ---- GAT layer 1 ----
    {
        int tiles = (NNODES / 16) * (F1 / 16);   // 1875 * 16 = 30000
        k_gemm_wmma_f16<EMB><<<CDIV(tiles, B / 32), B, 0, stream>>>(x16, w1t, h1, F1, tiles);
    }
    k_att_dots<<<CDIV(NNODES * NH1, B), B, 0, stream>>>(h1, as1, ad1, als1, ald1, NH1);
    k_edge_lrelu_max<<<CDIV(ETOT * NH1, B), B, 0, stream>>>(src, dst, als1, ald1, e1, m1, NH1);
    k_edge_expsum<<<CDIV(ETOT * NH1, B), B, 0, stream>>>(dst, e1, m1, s1, NH1);
    k_edge_aggregate<<<CDIV(ETOT * NH1, B), B, 0, stream>>>(src, dst, e1, s1, h1, out1, NH1);
    k_bias_elu_to_f16<<<CDIV(NNODES * F1, B), B, 0, stream>>>(out1, b1, x1h, F1, NNODES * F1);

    // ---- GAT layer 2 ----
    {
        int tiles = (NNODES / 16) * (HID / 16);  // 1875 * 2 = 3750
        k_gemm_wmma_f16<F1><<<CDIV(tiles, B / 32), B, 0, stream>>>(x1h, w2t, h2, HID, tiles);
    }
    k_att_dots<<<CDIV(NNODES, B), B, 0, stream>>>(h2, as2, ad2, als2, ald2, 1);
    k_edge_lrelu_max<<<CDIV(ETOT, B), B, 0, stream>>>(src, dst, als2, ald2, e2, m2, 1);
    k_edge_expsum<<<CDIV(ETOT, B), B, 0, stream>>>(dst, e2, m2, s2, 1);
    k_edge_aggregate<<<CDIV(ETOT, B), B, 0, stream>>>(src, dst, e2, s2, h2, out2, 1);
    k_bias_elu_f32<<<CDIV(NNODES * HID, B), B, 0, stream>>>(out2, b2, x2, HID, NNODES * HID);

    // ---- FC + MDN head ----
    k_head<<<CDIV(NNODES, B), B, 0, stream>>>(x2, Wfc, bfc, Wmu, bmu, Wvar, bvar, Wpi, bpi, out);
}
